// Fleet_Decoder_40364102648304
// MI455X (gfx1250) — matrix operand order repaired and checked by reading, not compiled, not run on hardware
//
#include <hip/hip_runtime.h>

typedef __bf16 bf16_t;
typedef __attribute__((ext_vector_type(16))) __bf16 v16bf;
typedef __attribute__((ext_vector_type(8)))  float  v8f;

static constexpr int BB = 32;
static constexpr int PP = 100;
static constexpr int NN = 5000;
static constexpr int MM = 5001;
static constexpr int AA = 50;
static constexpr int DD = 128;
static constexpr int COLS = 144;      // 128 enc cols + 16 stat cols
static constexpr int PROWS = 112;     // 7 tiles of 16 (P=100 padded)
static constexpr int NCH = 10;        // M chunks (512 each)
static constexpr int MCHUNK = 512;
static constexpr float INV_SQRT_D = 0.08838834764831845f;

// ---------------------------------------------------------------- feat kernel
__global__ void feat_kernel(const float* __restrict__ xy,
                            const float* __restrict__ dem,
                            float* __restrict__ Fmat)
{
  int gid = blockIdx.x * blockDim.x + threadIdx.x;
  if (gid >= BB * MM) return;
  int b = gid / MM, m = gid % MM;
  float f[16];
  #pragma unroll
  for (int i = 0; i < 16; ++i) f[i] = 0.f;
  f[0] = 1.f;                          // counts toward num_unv (all m incl depot)
  if (m > 0) {
    int n = m - 1;
    float x = xy[((size_t)b * NN + n) * 2 + 0];
    float y = xy[((size_t)b * NN + n) * 2 + 1];
    float d = dem[(size_t)b * NN + n];
    f[1] = 1.f;           // nu (nodes only)
    f[2] = d;             // rem
    f[3] = x;  f[4] = y;
    f[5] = x * x;  f[6] = y * y;
    f[7] = d * x;  f[8] = d * y;
    f[9] = d * (x * x + y * y);
  }
  float* o = Fmat + (size_t)gid * 16;
  #pragma unroll
  for (int i = 0; i < 16; ++i) o[i] = f[i];
}

// ---------------------------------------------------------------- agent path
__device__ __forceinline__ void gemm50(const float* __restrict__ in,
                                       const float* __restrict__ W,
                                       const float* __restrict__ bias,
                                       float* __restrict__ outp,
                                       int t, bool relu)
{
  int d = t & 127;
  for (int a = (t >> 7); a < AA; a += 2) {
    float acc = bias ? bias[d] : 0.f;
    const float* row = in + a * DD;
    #pragma unroll 8
    for (int k = 0; k < DD; ++k) acc = fmaf(row[k], W[k * DD + d], acc);
    if (relu) acc = fmaxf(acc, 0.f);
    outp[a * DD + d] = acc;
  }
}

__global__ __launch_bounds__(256) void agents_kernel(
    const float* __restrict__ cap,  const float* __restrict__ fixc,
    const float* __restrict__ varc,
    const float* __restrict__ embW, const float* __restrict__ embB,
    const float* __restrict__ qW,   const float* __restrict__ kW,
    const float* __restrict__ vW,
    const float* __restrict__ combW,const float* __restrict__ combB,
    const float* __restrict__ lnG,  const float* __restrict__ lnB,
    const float* __restrict__ WkW,  const float* __restrict__ WkB,
    const float* __restrict__ m1W,  const float* __restrict__ m1B,
    const float* __restrict__ m2W,  const float* __restrict__ m2B,
    const float* __restrict__ node_demand,
    float* __restrict__ kOut, float* __restrict__ fcnOut,
    float* __restrict__ totdOut)
{
  const int b = blockIdx.x, t = threadIdx.x;
  __shared__ float bufA[AA*DD], bufQ[AA*DD], bufK[AA*DD], bufV[AA*DD];
  __shared__ float attn[AA*52];
  __shared__ float fcv[AA], rowMu[AA], rowIS[AA];
  __shared__ float red[256];
  __shared__ float scal[2];

  // total demand per batch
  float s = 0.f;
  for (int n = t; n < NN; n += 256) s += node_demand[(size_t)b * NN + n];
  red[t] = s; __syncthreads();
  for (int off = 128; off > 0; off >>= 1) { if (t < off) red[t] += red[t + off]; __syncthreads(); }
  if (t == 0) totdOut[b] = red[0];

  const float* capb = cap + b * AA;
  const float* fixb = fixc + b * AA;
  const float* vab  = varc + b * AA;
  for (int idx = t; idx < AA * DD; idx += 256) {
    int a = idx >> 7, d = idx & 127;
    bufA[idx] = fmaf(capb[a], embW[d],
                fmaf(fixb[a], embW[DD + d],
                fmaf(vab[a],  embW[2 * DD + d], embB[d])));
  }
  __syncthreads();
  gemm50(bufA, qW, nullptr, bufQ, t, false);
  gemm50(bufA, kW, nullptr, bufK, t, false);
  gemm50(bufA, vW, nullptr, bufV, t, false);
  __syncthreads();
  for (int idx = t; idx < AA * AA; idx += 256) {
    int a = idx / AA, e = idx % AA;
    float sd = 0.f;
    #pragma unroll 8
    for (int d = 0; d < DD; ++d) sd = fmaf(bufQ[a*DD+d], bufK[e*DD+d], sd);
    attn[a*52 + e] = sd * INV_SQRT_D;
  }
  __syncthreads();
  if (t < AA) {
    float mx = -1e30f;
    for (int e = 0; e < AA; ++e) mx = fmaxf(mx, attn[t*52+e]);
    float sm = 0.f;
    for (int e = 0; e < AA; ++e) { float ex = __expf(attn[t*52+e] - mx); attn[t*52+e] = ex; sm += ex; }
    float inv = 1.f / sm;
    for (int e = 0; e < AA; ++e) attn[t*52+e] *= inv;
  }
  __syncthreads();
  { // ctx -> bufQ (qv dead)
    int d = t & 127;
    for (int a = (t >> 7); a < AA; a += 2) {
      float acc = 0.f;
      for (int e = 0; e < AA; ++e) acc = fmaf(attn[a*52+e], bufV[e*DD+d], acc);
      bufQ[a*DD+d] = acc;
    }
  }
  __syncthreads();
  gemm50(bufQ, combW, combB, bufK, t, false);   // x (partial) -> bufK (kv dead)
  __syncthreads();
  for (int idx = t; idx < AA * DD; idx += 256) bufK[idx] += bufA[idx];  // x = ae + ...
  __syncthreads();
  if (t < AA) {
    float mu = 0.f;
    for (int d = 0; d < DD; ++d) mu += bufK[t*DD+d];
    mu *= (1.f / DD);
    float vr = 0.f;
    for (int d = 0; d < DD; ++d) { float dv = bufK[t*DD+d] - mu; vr = fmaf(dv, dv, vr); }
    vr *= (1.f / DD);
    rowMu[t] = mu; rowIS[t] = rsqrtf(vr + 1e-5f);
  }
  __syncthreads();
  for (int idx = t; idx < AA * DD; idx += 256) {
    int a = idx >> 7, d = idx & 127;
    bufV[idx] = fmaf((bufK[idx] - rowMu[a]) * rowIS[a], lnG[d], lnB[d]);  // agent_emb
  }
  __syncthreads();
  gemm50(bufV, WkW, WkB, kOut + (size_t)b * AA * DD, t, false);
  gemm50(bufV, m1W, m1B, bufA, t, true);
  __syncthreads();
  if (t < AA) {
    float fc = m2B[0];
    #pragma unroll 8
    for (int d = 0; d < DD; ++d) fc = fmaf(bufA[t*DD+d], m2W[d], fc);
    fcv[t] = fc;
  }
  __syncthreads();
  if (t == 0) {
    float m = 0.f;
    for (int a = 0; a < AA; ++a) m += fcv[a];
    m *= (1.f / AA);
    float v2 = 0.f;
    for (int a = 0; a < AA; ++a) { float dv = fcv[a] - m; v2 = fmaf(dv, dv, v2); }
    float sd = sqrtf(v2 / (AA - 1)) + 1e-6f;
    scal[0] = m; scal[1] = 1.f / sd;
  }
  __syncthreads();
  if (t < AA) fcnOut[b * AA + t] = (fcv[t] - scal[0]) * scal[1];
}

// ---------------------------------------------------------------- WMMA GEMM
union FragBF { v16bf v; bf16_t e[16]; };

__global__ __launch_bounds__(288) void gctx_kernel(
    const float* __restrict__ enc, const float* __restrict__ Fmat,
    const int* __restrict__ vis, float* __restrict__ Gpart)
{
  const int b = blockIdx.y, chunk = blockIdx.x, t = threadIdx.x;
  const int wave = t >> 5, lane = t & 31;
  const int nIdx = lane & 15, hf = lane >> 4;
  __shared__ bf16_t ldsA[PROWS * 40];   // unv tile, stride 40 (bank-conflict free)

  v8f acc[7];
  #pragma unroll
  for (int pt = 0; pt < 7; ++pt) acc[pt] = v8f{};

  const float* Bsrc; int stride, col;
  if (wave < 8) { Bsrc = enc  + (size_t)b * MM * DD; stride = DD; col = wave * 16 + nIdx; }
  else          { Bsrc = Fmat + (size_t)b * MM * 16; stride = 16; col = nIdx; }

  const int mBase0 = chunk * MCHUNK;
  for (int ks = 0; ks < MCHUNK / 32; ++ks) {
    const int mb = mBase0 + ks * 32;
    __syncthreads();
    // stage unv (bf16, exact 0/1) into LDS
    for (int idx = t; idx < PROWS * 32; idx += 288) {
      int p = idx >> 5, ml = idx & 31, m = mb + ml;
      int v = (p < PP && m < MM) ? vis[((size_t)b * PP + p) * MM + m] : 1;
      ldsA[p * 40 + ml] = (bf16_t)(float)(1 - v);
    }
    __syncthreads();
    // B fragment: 32x16 chunk, split into bf16 hi + lo for ~fp32 accuracy
    FragBF bh, bl;
    #pragma unroll
    for (int i = 0; i < 16; ++i) {
      int k = (i < 8) ? (hf * 8 + i) : (16 + hf * 8 + (i - 8));
      int m = mb + k;
      float f = (m < MM) ? Bsrc[(size_t)m * stride + col] : 0.f;
      bf16_t h16 = (bf16_t)f;
      bh.e[i] = h16;
      bl.e[i] = (bf16_t)(f - (float)h16);
    }
    #pragma unroll
    for (int pt = 0; pt < 7; ++pt) {
      FragBF af;
      const bf16_t* row = &ldsA[(pt * 16 + nIdx) * 40];
      const int off = hf * 8;
      #pragma unroll
      for (int i = 0; i < 8; ++i) { af.e[i] = row[off + i]; af.e[8 + i] = row[16 + off + i]; }
      acc[pt] = __builtin_amdgcn_wmma_f32_16x16x32_bf16(false, af.v, false, bh.v,
                                                        (short)0, acc[pt], false, false);
      acc[pt] = __builtin_amdgcn_wmma_f32_16x16x32_bf16(false, af.v, false, bl.v,
                                                        (short)0, acc[pt], false, false);
    }
  }
  // store partial C tiles
  const int colOut = (wave < 8 ? wave * 16 : 128) + nIdx;
  float* gp = Gpart + ((size_t)b * NCH + chunk) * PROWS * COLS;
  #pragma unroll
  for (int pt = 0; pt < 7; ++pt) {
    #pragma unroll
    for (int j = 0; j < 8; ++j) {
      int p = pt * 16 + j + 8 * hf;
      gp[(size_t)p * COLS + colOut] = acc[pt][j];
    }
  }
}

// ---------------------------------------------------------------- finalize
__global__ __launch_bounds__(256) void finalize_kernel(
    const float* __restrict__ enc,  const float* __restrict__ Gpart,
    const float* __restrict__ WqW,  const float* __restrict__ WqB,
    const float* __restrict__ kBuf, const float* __restrict__ fcnBuf,
    const float* __restrict__ totd, const float* __restrict__ fleet,
    float* __restrict__ out)
{
  const int b = blockIdx.y, pt = blockIdx.x, t = threadIdx.x;
  __shared__ float WqL[260 * 128];    // 130 KB, reused for 16 policies
  __shared__ float kL[AA * 132];      // padded stride, conflict-free
  __shared__ float fcnL[AA], depotL[DD];
  __shared__ float gsum[COLS], fqL[260], qL[DD], logit[52], sc[4];

  for (int idx = t; idx < 260 * 128; idx += 256) WqL[idx] = WqW[idx];
  for (int idx = t; idx < AA * DD; idx += 256) {
    int a = idx >> 7, d = idx & 127;
    kL[a * 132 + d] = kBuf[(size_t)b * AA * DD + idx];
  }
  if (t < AA) fcnL[t] = fcnBuf[b * AA + t];
  if (t < DD) depotL[t] = enc[(size_t)b * MM * DD + t];
  __syncthreads();
  const float totD = totd[b];

  for (int pr = 0; pr < 16; ++pr) {
    int p = pt * 16 + pr;
    if (p >= PP) break;
    if (t < COLS) {
      float s2 = 0.f;
      const float* gp = Gpart + ((size_t)b * NCH * PROWS + p) * COLS + t;
      #pragma unroll
      for (int c = 0; c < NCH; ++c) s2 += gp[(size_t)c * PROWS * COLS];
      gsum[t] = s2;
    }
    __syncthreads();
    if (t == 0) {
      float invNum = 1.f / (gsum[128] + 1e-8f);
      sc[0] = invNum;
      float nu  = fmaxf(gsum[129], 1.f);
      float rem = gsum[130];
      float mx = gsum[131] / nu, my = gsum[132] / nu;
      float vx = gsum[133] / nu - mx * mx;
      float vy = gsum[134] / nu - my * my;
      float wnum = gsum[137] - 2.f * (mx * gsum[135] + my * gsum[136])
                 + (mx * mx + my * my) * rem;
      float wvar = wnum / fmaxf(rem, 1e-8f);
      fqL[256] = rem / (totD + 1e-8f);
      fqL[257] = sqrtf(fmaxf(vx, 1e-8f));
      fqL[258] = sqrtf(fmaxf(vy, 1e-8f));
      fqL[259] = sqrtf(fmaxf(wvar, 1e-8f));
    }
    __syncthreads();
    if (t < DD) { fqL[t] = depotL[t]; fqL[DD + t] = gsum[t] * sc[0]; }
    __syncthreads();
    if (t < DD) {
      float qv = WqB[t];
      #pragma unroll 4
      for (int j = 0; j < 260; ++j) qv = fmaf(fqL[j], WqL[j * 128 + t], qv);
      qL[t] = qv;
    }
    __syncthreads();
    if (t < AA) {
      float sd = 0.f;
      #pragma unroll 8
      for (int d = 0; d < DD; ++d) sd = fmaf(qL[d], kL[t * 132 + d], sd);
      logit[t] = sd * INV_SQRT_D - fcnL[t] + fleet[((size_t)b * PP + p) * AA + t];
    }
    __syncthreads();
    if (t == 0) {
      float mx2 = -1e30f;
      for (int a = 0; a < AA; ++a) mx2 = fmaxf(mx2, logit[a]);
      float sm = 0.f;
      for (int a = 0; a < AA; ++a) sm += __expf(logit[a] - mx2);
      sc[1] = mx2; sc[2] = 1.f / sm;
    }
    __syncthreads();
    if (t < AA) out[((size_t)b * PP + p) * AA + t] = __expf(logit[t] - sc[1]) * sc[2];
    __syncthreads();
  }
}

// ---------------------------------------------------------------- launch
extern "C" void kernel_launch(void* const* d_in, const int* in_sizes, int n_in,
                              void* d_out, int out_size, void* d_ws, size_t ws_size,
                              hipStream_t stream)
{
  const float* enc   = (const float*)d_in[0];
  const float* xy    = (const float*)d_in[1];
  const float* dem   = (const float*)d_in[2];
  const float* cap   = (const float*)d_in[3];
  const float* fixc  = (const float*)d_in[4];
  const float* varc  = (const float*)d_in[5];
  const float* fleet = (const float*)d_in[6];
  const float* embW  = (const float*)d_in[7];
  const float* embB  = (const float*)d_in[8];
  const float* qW    = (const float*)d_in[9];
  const float* kW    = (const float*)d_in[10];
  const float* vW    = (const float*)d_in[11];
  const float* combW = (const float*)d_in[12];
  const float* combB = (const float*)d_in[13];
  const float* lnG   = (const float*)d_in[14];
  const float* lnB   = (const float*)d_in[15];
  const float* WqW   = (const float*)d_in[16];
  const float* WqB   = (const float*)d_in[17];
  const float* WkW   = (const float*)d_in[18];
  const float* WkB   = (const float*)d_in[19];
  const float* m1W   = (const float*)d_in[20];
  const float* m1B   = (const float*)d_in[21];
  const float* m2W   = (const float*)d_in[22];
  const float* m2B   = (const float*)d_in[23];
  const int*   vis   = (const int*)d_in[24];
  float* outF = (float*)d_out;

  float* ws = (float*)d_ws;
  float* Fmat   = ws;                                      // 32*5001*16
  float* Gpart  = Fmat  + (size_t)BB * MM * 16;            // 32*10*112*144
  float* kBuf   = Gpart + (size_t)BB * NCH * PROWS * COLS; // 32*50*128
  float* fcnBuf = kBuf  + (size_t)BB * AA * DD;            // 32*50
  float* totd   = fcnBuf + (size_t)BB * AA;                // 32

  feat_kernel<<<dim3((BB * MM + 255) / 256), 256, 0, stream>>>(xy, dem, Fmat);
  agents_kernel<<<dim3(BB), 256, 0, stream>>>(cap, fixc, varc, embW, embB,
      qW, kW, vW, combW, combB, lnG, lnB, WkW, WkB, m1W, m1B, m2W, m2B,
      dem, kBuf, fcnBuf, totd);
  gctx_kernel<<<dim3(NCH, BB), 288, 0, stream>>>(enc, Fmat, vis, Gpart);
  finalize_kernel<<<dim3(7, BB), 256, 0, stream>>>(enc, Gpart, WqW, WqB,
      kBuf, fcnBuf, totd, fleet, outF);
}
